// DeepseekV3Attention_29506425323546
// MI455X (gfx1250) — matrix operand order repaired and checked by reading, not compiled, not run on hardware
//
#include <hip/hip_runtime.h>

// ---------------------------------------------------------------------------
// Types for CDNA5 WMMA (wave32, v_wmma_f32_16x16x32_bf16)
// ---------------------------------------------------------------------------
typedef __attribute__((ext_vector_type(16))) __bf16 v16bf;
typedef __attribute__((ext_vector_type(8)))  float  v8f;
typedef __attribute__((ext_vector_type(2)))  unsigned int v2u;
typedef __attribute__((ext_vector_type(4)))  unsigned int v4u;
typedef __attribute__((ext_vector_type(2)))  int v2i_t;

union AB {           // one 16x32 (A) or 32x16 (B) bf16 WMMA operand: 8 VGPRs
  v16bf bf;
  v2u   u2[4];
  v4u   u4[2];
};

// CDNA5 async global->LDS path (ASYNCcnt-tracked, bypasses VGPRs).
// Probe-confirmed signature: (v2i AS1*, v2i AS3*, int offset, int cpol)
#if __has_builtin(__builtin_amdgcn_global_load_async_to_lds_b64) && \
    __has_builtin(__builtin_amdgcn_s_wait_asynccnt)
#define HAVE_ASYNC_LDS 1
#define AS1 __attribute__((address_space(1)))
#define AS3 __attribute__((address_space(3)))
__device__ __forceinline__ void async_b64(const unsigned short* g,
                                          unsigned short* l) {
  __builtin_amdgcn_global_load_async_to_lds_b64(
      (AS1 v2i_t*)(unsigned short*)g, (AS3 v2i_t*)l, 0, 0);
}
__device__ __forceinline__ void async_wait0() {
  __builtin_amdgcn_s_wait_asynccnt(0);
}
__device__ __forceinline__ void async_wait8() {
  __builtin_amdgcn_s_wait_asynccnt(8);
}
#else
#define HAVE_ASYNC_LDS 0
__device__ __forceinline__ void async_wait0() {}
__device__ __forceinline__ void async_wait8() {}
#endif

__device__ __forceinline__ unsigned short f2bf(float f) {
  unsigned u = __float_as_uint(f);
  u += 0x7FFFu + ((u >> 16) & 1u);          // round-to-nearest-even
  return (unsigned short)(u >> 16);
}
__device__ __forceinline__ float bf2f(unsigned short s) {
  return __uint_as_float(((unsigned)s) << 16);
}
__device__ __forceinline__ v8f v8f_zero() {
  v8f z = {0.f,0.f,0.f,0.f,0.f,0.f,0.f,0.f};
  return z;
}
__device__ __forceinline__ v8f wmma_bf16(const AB& a, const AB& b, v8f c) {
  return __builtin_amdgcn_wmma_f32_16x16x32_bf16(false, a.bf, false, b.bf,
                                                 (short)0, c, false, false);
}

// ---------------------------------------------------------------------------
// fp32 -> bf16 conversion (grid-stride)
// ---------------------------------------------------------------------------
__global__ __launch_bounds__(256) void f32_to_bf16_kernel(
    const float* __restrict__ src, unsigned short* __restrict__ dst, int n) {
  int i = blockIdx.x * blockDim.x + threadIdx.x;
  int stride = gridDim.x * blockDim.x;
  for (; i < n; i += stride) dst[i] = f2bf(src[i]);
}

// ---------------------------------------------------------------------------
// Generic bf16 GEMM: C[M,N] = A[M,K] @ B[N,K]^T, f32 accumulate.
// Block tile 128x128, 256 threads = 8 waves, each wave 32(M)x64(N).
// Double-buffered LDS staging pipelined over the async global->LDS engine:
// stage k+1's 8 loads/wave are issued before waiting for stage k's 8
// (async loads complete in order per wave => s_wait_asynccnt 8 suffices).
// Ragged N handled by clamping OOB B rows (results land in epilogue-guarded
// columns only, never stored).
// ---------------------------------------------------------------------------
#define LTS 40   // LDS row stride in bf16 elems (80B: 8B-aligned, conflict-free)

template <typename OutT>
__global__ __launch_bounds__(256) void gemm_bf16_tn(
    const unsigned short* __restrict__ A,
    const unsigned short* __restrict__ B,
    OutT* __restrict__ C,
    int M, int N, int K)
{
  __shared__ __align__(16) unsigned short As[2][128 * LTS];
  __shared__ __align__(16) unsigned short Bs[2][128 * LTS];

  const int tid = threadIdx.x;
  const int l   = tid & 31, wid = tid >> 5;
  const int wm  = wid & 3,  wn  = wid >> 2;     // 4x2 wave grid
  const int gm0 = blockIdx.y * 128, gn0 = blockIdx.x * 128;
  const int m16 = l & 15, hi = l >> 4;
  const int kb8 = hi * 8, kb16 = hi * 16;

  v8f acc[2][4];
#pragma unroll
  for (int i = 0; i < 2; ++i)
#pragma unroll
    for (int j = 0; j < 4; ++j) acc[i][j] = v8f_zero();

  // Stage one 128x32 A tile + 128x32 B tile into LDS buffer `buf`.
  auto stage = [&](int kc, int buf) {
#pragma unroll
    for (int it = 0; it < 4; ++it) {          // A: 1024 x 8B
      int i = tid + it * 256;
      int row = i >> 3, c = i & 7;
      const unsigned short* gp = A + (size_t)(gm0 + row) * K + kc + c * 4;
#if HAVE_ASYNC_LDS
      async_b64(gp, As[buf] + row * LTS + c * 4);
#else
      *(v2u*)(As[buf] + row * LTS + c * 4) = *(const v2u*)gp;
#endif
      if (kc + 64 < K) __builtin_prefetch(gp + 64, 0, 1);
    }
#pragma unroll
    for (int it = 0; it < 4; ++it) {          // B: rows clamped for ragged N
      int i = tid + it * 256;
      int row = i >> 3, c = i & 7;
      int gn = gn0 + row;
      if (gn >= N) gn = N - 1;
      const unsigned short* gp = B + (size_t)gn * K + kc + c * 4;
#if HAVE_ASYNC_LDS
      async_b64(gp, Bs[buf] + row * LTS + c * 4);
#else
      *(v2u*)(Bs[buf] + row * LTS + c * 4) = *(const v2u*)gp;
#endif
      if (kc + 64 < K) __builtin_prefetch(gp + 64, 0, 1);
    }
  };

  const int nk = K >> 5;
  stage(0, 0);
  for (int i = 0; i < nk; ++i) {
    const int cur = i & 1;
    if (i + 1 < nk) {
      stage((i + 1) * 32, cur ^ 1);           // overlap next stage with wait
      async_wait8();                          // current stage's 8 are older
    } else {
      async_wait0();
    }
    __syncthreads();

    AB a[2], b[4];
#pragma unroll
    for (int mt = 0; mt < 2; ++mt) {
      const unsigned short* p = As[cur] + (wm * 32 + mt * 16 + m16) * LTS;
      a[mt].u2[0] = *(const v2u*)(p + kb16);
      a[mt].u2[1] = *(const v2u*)(p + kb16 + 4);
      a[mt].u2[2] = *(const v2u*)(p + kb16 + 8);
      a[mt].u2[3] = *(const v2u*)(p + kb16 + 12);
    }
#pragma unroll
    for (int nt = 0; nt < 4; ++nt) {
      const unsigned short* p =
          Bs[cur] + (wn * 64 + nt * 16 + m16) * LTS + kb16;
      b[nt].u2[0] = *(const v2u*)(p);
      b[nt].u2[1] = *(const v2u*)(p + 4);
      b[nt].u2[2] = *(const v2u*)(p + 8);
      b[nt].u2[3] = *(const v2u*)(p + 12);
    }
#pragma unroll
    for (int mt = 0; mt < 2; ++mt)
#pragma unroll
      for (int nt = 0; nt < 4; ++nt)
        acc[mt][nt] = wmma_bf16(a[mt], b[nt], acc[mt][nt]);
    __syncthreads();
  }

  // Epilogue: C/D layout => row = base + 8*hi + r, col = base + (l&15)
#pragma unroll
  for (int mt = 0; mt < 2; ++mt)
#pragma unroll
    for (int nt = 0; nt < 4; ++nt)
#pragma unroll
      for (int r = 0; r < 8; ++r) {
        int row = gm0 + wm * 32 + mt * 16 + hi * 8 + r;
        int col = gn0 + wn * 64 + nt * 16 + m16;
        if (col < N) {
          float v = acc[mt][nt][r];
          if constexpr (sizeof(OutT) == 2)
            C[(size_t)row * N + col] = (OutT)f2bf(v);
          else
            C[(size_t)row * N + col] = (OutT)v;
        }
      }
}

// ---------------------------------------------------------------------------
// RMSNorm (fp32 in, bf16 out). One block per row.
// ---------------------------------------------------------------------------
__global__ __launch_bounds__(256) void rmsnorm_kernel(
    const float* __restrict__ x, const float* __restrict__ w,
    unsigned short* __restrict__ out, int n, int xstride, int ostride)
{
  __shared__ float red[8];
  const int row = blockIdx.x;
  const float* xr = x + (size_t)row * xstride;
  float ss = 0.f;
  for (int i = threadIdx.x; i < n; i += 256) {
    float v = xr[i];
    ss += v * v;
  }
#pragma unroll
  for (int m = 16; m >= 1; m >>= 1) ss += __shfl_xor(ss, m, 32);
  if ((threadIdx.x & 31) == 0) red[threadIdx.x >> 5] = ss;
  __syncthreads();
  if (threadIdx.x < 8) {
    float v = red[threadIdx.x];
#pragma unroll
    for (int m = 4; m >= 1; m >>= 1) v += __shfl_xor(v, m, 32);
    if (threadIdx.x == 0) red[0] = v;
  }
  __syncthreads();
  float rs = rsqrtf(red[0] / (float)n + 1e-6f);
  unsigned short* orow = out + (size_t)row * ostride;
  for (int i = threadIdx.x; i < n; i += 256)
    orow[i] = f2bf(w[i] * xr[i] * rs);
}

// ---------------------------------------------------------------------------
// RoPE for Q (in-place on bf16 q buffer [BS, 16*192], rope dims at h*192+128)
// ---------------------------------------------------------------------------
#define LN_THETA 9.210340371976184f   // ln(10000)

__global__ __launch_bounds__(256) void rope_q_kernel(
    unsigned short* __restrict__ q, int S)
{
  const int row = blockIdx.x;
  const float pos = (float)(row % S);
#pragma unroll
  for (int it = 0; it < 2; ++it) {
    int idx = threadIdx.x + it * 256;
    int h = idx >> 5, i = idx & 31;
    size_t base = (size_t)row * 3072 + h * 192 + 128;
    float inv = __expf(-((float)(2 * i) / 64.f) * LN_THETA);
    float ang = pos * inv;
    float c = __cosf(ang), s = __sinf(ang);
    float x1 = bf2f(q[base + i]);
    float x2 = bf2f(q[base + i + 32]);
    q[base + i]      = f2bf(x1 * c - x2 * s);
    q[base + i + 32] = f2bf(x2 * c + x1 * s);
  }
}

// RoPE for K (fp32 kv_a[...,512:576] -> bf16 krope[BS,64])
__global__ void rope_k_kernel(const float* __restrict__ kva,
                              unsigned short* __restrict__ krope, int S)
{
  const int row = blockIdx.x;
  const int i = threadIdx.x;            // 32 threads
  const float pos = (float)(row % S);
  float inv = __expf(-((float)(2 * i) / 64.f) * LN_THETA);
  float ang = pos * inv;
  float c = __cosf(ang), s = __sinf(ang);
  float x1 = kva[(size_t)row * 576 + 512 + i];
  float x2 = kva[(size_t)row * 576 + 512 + 32 + i];
  krope[(size_t)row * 64 + i]      = f2bf(x1 * c - x2 * s);
  krope[(size_t)row * 64 + 32 + i] = f2bf(x2 * c + x1 * s);
}

// ---------------------------------------------------------------------------
// MLA flash attention (causal, online softmax).
// Block = 256 threads = 8 waves, one (b,h,128-row q-tile) per block.
// Each wave owns 16 q rows. K-blocks of 32 keys streamed through LDS.
// B operands double-buffered in registers so LDS loads for step j+1 issue
// before the WMMA of step j (avoids s_wait_dscnt 0 per WMMA).
// ---------------------------------------------------------------------------
#define ATT_SCALE 0.07216878364870323f   // 192^-0.5

__global__ __launch_bounds__(256) void mla_flash_kernel(
    const unsigned short* __restrict__ Q,
    const unsigned short* __restrict__ KV,
    const unsigned short* __restrict__ KR,
    unsigned short* __restrict__ O,
    int S)
{
  constexpr int KT   = 200;  // Ks row stride (elems)
  constexpr int VT_S = 40;   // Vt row stride
  constexpr int PT_S = 40;   // P  row stride
  __shared__ __align__(16) unsigned short Ks[32 * KT];        // 32 keys x 192
  __shared__ __align__(16) unsigned short Vt[128 * VT_S];     // transposed V
  __shared__ __align__(16) unsigned short Pl[8 * 16 * PT_S];  // per-wave P

  const int nqt = S >> 7;
  const int qt = blockIdx.x % nqt;
  const int h  = (blockIdx.x / nqt) % 16;
  const int b  = blockIdx.x / (nqt * 16);
  const int tid = threadIdx.x;
  const int l = tid & 31, w = tid >> 5;
  const int m16 = l & 15, hi = l >> 4;
  const int kb8 = hi * 8, kb16 = hi * 16;
  const size_t bS = (size_t)b * S;

  // --- load this wave's 16x192 Q stripe into registers (A-operand layout) ---
  AB qa[6];
  {
    const unsigned short* qr =
        Q + (bS + qt * 128 + w * 16 + m16) * 3072 + h * 192;
#pragma unroll
    for (int fc = 0; fc < 6; ++fc) {
      qa[fc].u4[0] = *(const v4u*)(qr + fc * 32 + kb16);
      qa[fc].u4[1] = *(const v4u*)(qr + fc * 32 + kb16 + 8);
    }
  }

  v8f oacc[8];
  float mrow[8], lrow[8];
#pragma unroll
  for (int i = 0; i < 8; ++i) {
    oacc[i] = v8f_zero();
    mrow[i] = -3.0e38f;
    lrow[i] = 0.f;
  }

  const int nkb = qt * 4 + 4;   // causal: keys up to qt*128+127
  for (int ib = 0; ib < nkb; ++ib) {
    const int kb = ib * 32;
    __syncthreads();
    // K nope: 32 keys x 128 feats (8B granularity) -> async to LDS if avail
#pragma unroll
    for (int it = 0; it < 4; ++it) {
      int i = tid + it * 256;
      int key = i >> 5, c = i & 31;
      const unsigned short* gp =
          KV + (bS + kb + key) * 4096 + h * 256 + c * 4;
#if HAVE_ASYNC_LDS
      async_b64(gp, Ks + key * KT + c * 4);
#else
      *(v2u*)(Ks + key * KT + c * 4) = *(const v2u*)gp;
#endif
    }
    // K rope: 32 keys x 64 feats
#pragma unroll
    for (int it = 0; it < 2; ++it) {
      int i = tid + it * 256;
      int key = i >> 4, c = i & 15;
      const unsigned short* gp = KR + (bS + kb + key) * 64 + c * 4;
#if HAVE_ASYNC_LDS
      async_b64(gp, Ks + key * KT + 128 + c * 4);
#else
      *(v2u*)(Ks + key * KT + 128 + c * 4) = *(const v2u*)gp;
#endif
    }
    // V transposed: Vt[dv][key] (manual transpose, VGPR bounce)
#pragma unroll
    for (int it = 0; it < 4; ++it) {
      int i = tid + it * 256;
      int key = i >> 5, g = i & 31;
      v2u d = *(const v2u*)(KV + (bS + kb + key) * 4096 + h * 256 + 128 + g * 4);
      const unsigned short* e = (const unsigned short*)&d;
      Vt[(g * 4 + 0) * VT_S + key] = e[0];
      Vt[(g * 4 + 1) * VT_S + key] = e[1];
      Vt[(g * 4 + 2) * VT_S + key] = e[2];
      Vt[(g * 4 + 3) * VT_S + key] = e[3];
    }
    async_wait0();
    __syncthreads();

    // ---- S = Q @ K^T over 192 dims: 12 WMMA, double-buffered B operand ----
    v8f sacc[2] = {v8f_zero(), v8f_zero()};
    AB bm[2];
    {
      const unsigned short* kr0 = Ks + m16 * KT + kb16;   // fc=0, t=0
      bm[0].u2[0] = *(const v2u*)(kr0);
      bm[0].u2[1] = *(const v2u*)(kr0 + 4);
      bm[0].u2[2] = *(const v2u*)(kr0 + 8);
      bm[0].u2[3] = *(const v2u*)(kr0 + 12);
    }
#pragma unroll
    for (int j = 0; j < 12; ++j) {
      const int fc = j >> 1, t = j & 1;
      if (j + 1 < 12) {
        const int fn = (j + 1) >> 1, tn = (j + 1) & 1;
        const unsigned short* kr =
            Ks + (tn * 16 + m16) * KT + fn * 32 + kb16;
        AB& nb = bm[(j + 1) & 1];
        nb.u2[0] = *(const v2u*)(kr);
        nb.u2[1] = *(const v2u*)(kr + 4);
        nb.u2[2] = *(const v2u*)(kr + 8);
        nb.u2[3] = *(const v2u*)(kr + 12);
      }
      sacc[t] = wmma_bf16(qa[fc], bm[j & 1], sacc[t]);
    }

    // ---- online softmax (row = r + 8*hi within wave stripe) ----
    unsigned short* pw = Pl + w * 16 * PT_S;
#pragma unroll
    for (int r = 0; r < 8; ++r) {
      int qpos = qt * 128 + w * 16 + hi * 8 + r;
      float s0 = (kb + m16      <= qpos) ? sacc[0][r] * ATT_SCALE : -1e30f;
      float s1 = (kb + 16 + m16 <= qpos) ? sacc[1][r] * ATT_SCALE : -1e30f;
      float rm = fmaxf(s0, s1);
#pragma unroll
      for (int msk = 8; msk >= 1; msk >>= 1)
        rm = fmaxf(rm, __shfl_xor(rm, msk, 32));
      float mn = fmaxf(mrow[r], rm);
      float alpha = __expf(mrow[r] - mn);
      mrow[r] = mn;
      float p0 = __expf(s0 - mn), p1 = __expf(s1 - mn);
      float rs = p0 + p1;
#pragma unroll
      for (int msk = 8; msk >= 1; msk >>= 1) rs += __shfl_xor(rs, msk, 32);
      lrow[r] = lrow[r] * alpha + rs;
#pragma unroll
      for (int to = 0; to < 8; ++to) oacc[to][r] *= alpha;
      pw[(hi * 8 + r) * PT_S + m16]      = f2bf(p0);
      pw[(hi * 8 + r) * PT_S + 16 + m16] = f2bf(p1);
    }

    // ---- O += P @ V  (P from LDS in A layout; V double-buffered) ----
    AB pa;
    {
      const unsigned short* pr = pw + m16 * PT_S;
      pa.u2[0] = *(const v2u*)(pr + kb16);
      pa.u2[1] = *(const v2u*)(pr + kb16 + 4);
      pa.u2[2] = *(const v2u*)(pr + kb16 + 8);
      pa.u2[3] = *(const v2u*)(pr + kb16 + 12);
    }
    AB vb[2];
    {
      const unsigned short* vr0 = Vt + m16 * VT_S + kb16;   // to=0
      vb[0].u2[0] = *(const v2u*)(vr0);
      vb[0].u2[1] = *(const v2u*)(vr0 + 4);
      vb[0].u2[2] = *(const v2u*)(vr0 + 8);
      vb[0].u2[3] = *(const v2u*)(vr0 + 12);
    }
#pragma unroll
    for (int to = 0; to < 8; ++to) {
      if (to + 1 < 8) {
        const unsigned short* vr =
            Vt + ((to + 1) * 16 + m16) * VT_S + kb16;
        AB& nv = vb[(to + 1) & 1];
        nv.u2[0] = *(const v2u*)(vr);
        nv.u2[1] = *(const v2u*)(vr + 4);
        nv.u2[2] = *(const v2u*)(vr + 8);
        nv.u2[3] = *(const v2u*)(vr + 12);
      }
      oacc[to] = wmma_bf16(pa, vb[to & 1], oacc[to]);
    }
  }

  // ---- finalize: O / l, write bf16 attn_out[tok, h*128 + dv] ----
#pragma unroll
  for (int to = 0; to < 8; ++to) {
#pragma unroll
    for (int r = 0; r < 8; ++r) {
      float v = oacc[to][r] / lrow[r];
      size_t tok = bS + qt * 128 + w * 16 + hi * 8 + r;
      O[tok * 2048 + h * 128 + to * 16 + m16] = f2bf(v);
    }
  }
}

// ---------------------------------------------------------------------------
// Host orchestration
// ---------------------------------------------------------------------------
static inline int cdiv_i(int a, int b) { return (a + b - 1) / b; }

extern "C" void kernel_launch(void* const* d_in, const int* in_sizes, int n_in,
                              void* d_out, int out_size, void* d_ws, size_t ws_size,
                              hipStream_t stream)
{
  constexpr int B = 2, S = 2048, H = 2048, BS = B * S;
  constexpr int QLR = 1536, NH = 16, DQK = 192, DNOPE = 128, DV = 128, KVL = 512;
  constexpr int NQ = NH * DQK;            // 3072
  constexpr int NKVA = KVL + 64;          // 576
  constexpr int NKV = NH * (DNOPE + DV);  // 4096

  const float* hs   = (const float*)d_in[0];
  const float* qaw  = (const float*)d_in[1];
  const float* qaln = (const float*)d_in[2];
  const float* qbw  = (const float*)d_in[3];
  const float* kvaw = (const float*)d_in[4];
  const float* kvln = (const float*)d_in[5];
  const float* kvbw = (const float*)d_in[6];
  const float* ow   = (const float*)d_in[7];

  unsigned char* wsp = (unsigned char*)d_ws;
  size_t off = 0;
  auto take = [&](size_t elems, size_t esz) -> void* {
    void* p = wsp + off;
    off += (elems * esz + 255) & ~(size_t)255;
    return p;
  };
  unsigned short* h_bf    = (unsigned short*)take((size_t)BS * H, 2);
  unsigned short* qaw_bf  = (unsigned short*)take((size_t)QLR * H, 2);
  unsigned short* qbw_bf  = (unsigned short*)take((size_t)NQ * QLR, 2);
  unsigned short* kvaw_bf = (unsigned short*)take((size_t)NKVA * H, 2);
  unsigned short* kvbw_bf = (unsigned short*)take((size_t)NKV * KVL, 2);
  unsigned short* ow_bf   = (unsigned short*)take((size_t)H * H, 2);
  float*          qa_f    = (float*)take((size_t)BS * QLR, 4);
  float*          kva_f   = (float*)take((size_t)BS * NKVA, 4);
  unsigned short* qan_bf  = (unsigned short*)take((size_t)BS * QLR, 2);
  unsigned short* ckv_bf  = (unsigned short*)take((size_t)BS * KVL, 2);
  unsigned short* q_bf    = (unsigned short*)take((size_t)BS * NQ, 2);
  unsigned short* kv_bf   = (unsigned short*)take((size_t)BS * NKV, 2);
  unsigned short* kr_bf   = (unsigned short*)take((size_t)BS * 64, 2);
  unsigned short* attn_bf = (unsigned short*)take((size_t)BS * H, 2);

  // 1. convert fp32 weights/activations to bf16
  f32_to_bf16_kernel<<<1024, 256, 0, stream>>>(hs, h_bf, BS * H);
  f32_to_bf16_kernel<<<1024, 256, 0, stream>>>(qaw, qaw_bf, QLR * H);
  f32_to_bf16_kernel<<<1024, 256, 0, stream>>>(qbw, qbw_bf, NQ * QLR);
  f32_to_bf16_kernel<<<1024, 256, 0, stream>>>(kvaw, kvaw_bf, NKVA * H);
  f32_to_bf16_kernel<<<1024, 256, 0, stream>>>(kvbw, kvbw_bf, NKV * KVL);
  f32_to_bf16_kernel<<<1024, 256, 0, stream>>>(ow, ow_bf, H * H);

  // 2. q_a = hidden @ q_a_w^T ; kv_a = hidden @ kv_a_w^T   (fp32 out)
  {
    dim3 g1(cdiv_i(QLR, 128), BS / 128);
    gemm_bf16_tn<float><<<g1, 256, 0, stream>>>(h_bf, qaw_bf, qa_f, BS, QLR, H);
    dim3 g2(cdiv_i(NKVA, 128), BS / 128);
    gemm_bf16_tn<float><<<g2, 256, 0, stream>>>(h_bf, kvaw_bf, kva_f, BS, NKVA, H);
  }

  // 3. RMSNorm -> bf16
  rmsnorm_kernel<<<BS, 256, 0, stream>>>(qa_f, qaln, qan_bf, QLR, QLR, QLR);
  rmsnorm_kernel<<<BS, 256, 0, stream>>>(kva_f, kvln, ckv_bf, KVL, NKVA, KVL);

  // 4. q = q_a_n @ q_b_w^T ; kv = c_kv_n @ kv_b_w^T   (bf16 out)
  {
    dim3 g1(cdiv_i(NQ, 128), BS / 128);
    gemm_bf16_tn<unsigned short><<<g1, 256, 0, stream>>>(qan_bf, qbw_bf, q_bf, BS, NQ, QLR);
    dim3 g2(cdiv_i(NKV, 128), BS / 128);
    gemm_bf16_tn<unsigned short><<<g2, 256, 0, stream>>>(ckv_bf, kvbw_bf, kv_bf, BS, NKV, KVL);
  }

  // 5. RoPE
  rope_q_kernel<<<BS, 256, 0, stream>>>(q_bf, S);
  rope_k_kernel<<<BS, 32, 0, stream>>>(kva_f, kr_bf, S);

  // 6. flash attention -> attn_out bf16 [BS, NH*DV]
  mla_flash_kernel<<<B * NH * (S / 128), 256, 0, stream>>>(q_bf, kv_bf, kr_bf,
                                                           attn_bf, S);

  // 7. out = attn_out @ o_w^T   (fp32 to d_out)
  {
    dim3 g(cdiv_i(H, 128), BS / 128);
    gemm_bf16_tn<float><<<g, 256, 0, stream>>>(attn_bf, ow_bf, (float*)d_out,
                                               BS, H, NH * DV);
  }
}
